// ResLSTM_CRF_5841155523178
// MI455X (gfx1250) — matrix operand order repaired and checked by reading, not compiled, not run on hardware
//
#include <hip/hip_runtime.h>
#include <hip/hip_bf16.h>
#include <math.h>

// ---------------------------------------------------------------------------
// ResLSTM-CRF for MI455X (gfx1250, wave32, WMMA).
//  - fp16 WMMA (v_wmma_f32_16x16x32_f16) with fp32 accumulate everywhere.
//  - Batched input projections as tiled WMMA GEMMs (HBM-streamed, ~260MB).
//  - Persistent single-workgroup recurrence kernels: h in LDS (fp16, read as
//    A-fragments), c in VGPRs, gates live entirely in WMMA accumulators
//    (wave owns i/f/g/o columns for its slice -> no gate staging buffers).
//  - Anti-LICM: a zero offset is laundered through an empty asm each timestep
//    so weight loads can't be hoisted+spilled (round-1), while the base
//    pointer keeps its type so loads stay GLOBAL (round-2 showed FLAT loads,
//    which couple LOADcnt+DScnt and serialize against the LDS A-frag stream).
//  - Layer-2 recurrence processes the batch in two M-halves per step to keep
//    accumulators at 64 VGPRs/wave (no spills at 32 waves/WGP).
//  - Viterbi: one wave32 per batch row, backpointers in LDS.
// ---------------------------------------------------------------------------

typedef __attribute__((ext_vector_type(16))) _Float16 v16h;
typedef __attribute__((ext_vector_type(8)))  _Float16 v8h;
typedef __attribute__((ext_vector_type(8)))  float    v8f;

#define DEVINL __device__ __forceinline__

#define Bq   64
#define Sq   512
#define Eq   256
#define Hq   512
#define HHq  256
#define Tq   22
#define TPAD 32
#define START_TAG 20
#define STOP_TAG  21
#define RROWS (Sq * Bq)   // 32768 time-major rows (r = t*64 + b)

DEVINL v8f wmma_f16(v16h a, v16h b, v8f c) {
  // D = A(16x32 f16) x B(32x16 f16) + C(16x16 f32)
  return __builtin_amdgcn_wmma_f32_16x16x32_f16(false, a, false, b, (short)0, c,
                                                false, false);
}

// A-fragment: lane holds row m = lane%16; 16 contiguous f16 at
// k = kk + (lane<16?0:16) (matches B-fragment k order).
DEVINL v16h load_a_frag(const _Float16* p) {
  v8h lo = *(const v8h*)(p);
  v8h hi = *(const v8h*)(p + 8);
  return __builtin_shufflevector(lo, hi, 0, 1, 2, 3, 4, 5, 6, 7,
                                         8, 9, 10, 11, 12, 13, 14, 15);
}

DEVINL float sigm(float x) { return 1.0f / (1.0f + __expf(-x)); }

// ---------------------------------------------------------------------------
// small prep kernels
// ---------------------------------------------------------------------------
__global__ __launch_bounds__(256) void k_cvt(const float* __restrict__ s,
                                             _Float16* __restrict__ d, int n) {
  int i = blockIdx.x * 256 + threadIdx.x;
  if (i < n) d[i] = (_Float16)s[i];
}

__global__ __launch_bounds__(256) void k_wtag(const float* __restrict__ Wt,
                                              const float* __restrict__ bt,
                                              _Float16* __restrict__ wtag,
                                              float* __restrict__ btag) {
  int i = blockIdx.x * 256 + threadIdx.x;           // grid 64 -> 16384
  if (i < TPAD * Hq) {
    int row = i >> 9, col = i & (Hq - 1);
    wtag[i] = (row < Tq) ? (_Float16)Wt[row * Hq + col] : (_Float16)0.0f;
  }
  if (i < TPAD) btag[i] = (i < Tq) ? bt[i] : 0.0f;
}

__global__ __launch_bounds__(128) void k_gather(const int* __restrict__ sent,
                                                const float* __restrict__ emb,
                                                _Float16* __restrict__ x16) {
  int r = blockIdx.x;                    // r = t*64 + b  (time-major)
  int t = r >> 6, b = r & 63;
  int tok = sent[b * Sq + t];
  const float* src = emb + (long)tok * Eq;
  _Float16* dst = x16 + (long)r * Eq;
  int e = threadIdx.x * 2;
  dst[e]     = (_Float16)src[e];
  dst[e + 1] = (_Float16)src[e + 1];
}

// ---------------------------------------------------------------------------
// Batched GEMM: Out[R x N] = X_f16[R x K] @ W_f16[N x K]^T + bias[N]
// block = 256 threads (8 waves); block tile 128 rows x BN cols.
// wave w: m-tile w; BN/16 n-tiles. B[n][k] row-major == WMMA B layout.
// ---------------------------------------------------------------------------
template <int BN, bool OUTF32>
__global__ __launch_bounds__(256) void k_gemm(const _Float16* __restrict__ X,
                                              const _Float16* __restrict__ W,
                                              const float* __restrict__ bias,
                                              void* __restrict__ Out,
                                              int N, int K) {
  constexpr int NT = BN / 16;
  const int wave = threadIdx.x >> 5, lane = threadIdx.x & 31;
  const long mBase = (long)blockIdx.y * 128 + wave * 16;
  const int nBase = blockIdx.x * BN;
  const int selA = (lane & 16) ? 16 : 0;
  const int selB = (lane & 16) ? 16 : 0;
  v8f z = {0.f, 0.f, 0.f, 0.f, 0.f, 0.f, 0.f, 0.f};
  v8f acc[NT];
#pragma unroll
  for (int j = 0; j < NT; ++j) acc[j] = z;
  const _Float16* xrow = X + (mBase + (lane & 15)) * K;
  for (int kk = 0; kk < K; kk += 32) {
    v16h a = load_a_frag(xrow + kk + selA);
#pragma unroll
    for (int j = 0; j < NT; ++j) {
      const int n = nBase + j * 16 + (lane & 15);
      v16h bf = *(const v16h*)(W + (long)n * K + kk + selB);
      acc[j] = wmma_f16(a, bf, acc[j]);
    }
  }
#pragma unroll
  for (int j = 0; j < NT; ++j) {
#pragma unroll
    for (int r = 0; r < 8; ++r) {
      long row = mBase + ((lane & 16) ? r + 8 : r);
      int col = nBase + j * 16 + (lane & 15);
      float v = acc[j][r] + bias[col];
      if (OUTF32) ((float*)Out)[row * N + col] = v;
      else        ((_Float16*)Out)[row * N + col] = (_Float16)v;
    }
  }
}

// ---------------------------------------------------------------------------
// Layer-1 recurrence (bidirectional, HH=256). grid.x=2 (dir), 1024 thr.
// Gates C[64x1024] = G[t] + h@Whh^T. 256 WMMA tiles / 32 waves:
//   wave: column-group cg = w&15 (k in [cg*16,cg*16+16)), m-tiles {2(w>>4), +1}
//   n-tiles {cg, cg+16, cg+32, cg+48}  -> wave owns i,f,g,o for its columns.
// Cell update entirely in registers; h kept fp16 in LDS for next-step A-frags.
// ---------------------------------------------------------------------------
__global__ __launch_bounds__(1024) void k_lstm_l1(
    const _Float16* __restrict__ whhF, const _Float16* __restrict__ whhB,
    const _Float16* __restrict__ Gf, const _Float16* __restrict__ Gb,
    const float* __restrict__ h1_0, const float* __restrict__ c1_0,
    const int* __restrict__ lengths, _Float16* __restrict__ bi) {
  const int d = blockIdx.x;                       // 0=fwd, 1=bwd
  const _Float16* whhD = d ? whhB : whhF;
  const _Float16* G = d ? Gb : Gf;
  __shared__ _Float16 h_sh[Bq * HHq];             // 32 KB
  const int tid = threadIdx.x, wave = tid >> 5, lane = tid & 31;
  const int cg = wave & 15;
  const int mBase = (wave >> 4) * 2;
  const int kcol = cg * 16 + (lane & 15);
  const int selA = (lane & 16) ? 16 : 0;
  const int selB = (lane & 16) ? 16 : 0;
  for (int i = tid; i < Bq * HHq; i += 1024)
    h_sh[i] = (_Float16)h1_0[d * HHq + (i & (HHq - 1))];
  float c[2][8];
  int len[2][8];
#pragma unroll
  for (int mt = 0; mt < 2; ++mt)
#pragma unroll
    for (int r = 0; r < 8; ++r) {
      int b = (mBase + mt) * 16 + ((lane & 16) ? r + 8 : r);
      c[mt][r] = c1_0[d * HHq + kcol];
      len[mt][r] = lengths[b];
    }
  __syncthreads();
  v8f z = {0.f, 0.f, 0.f, 0.f, 0.f, 0.f, 0.f, 0.f};
  int inv = 0;   // always 0; opaquely redefined per step (anti-LICM)
  for (int s = 0; s < Sq; ++s) {
    const int t = d ? (Sq - 1 - s) : s;
    // Blocks LICM of the 32 loop-invariant B-fragments (would spill to
    // scratch) while keeping the typed base pointer -> GLOBAL loads.
    asm volatile("" : "+s"(inv));
    const _Float16* whh = whhD + inv;
    // prefetch next timestep's gate-stream row (HBM latency hiding)
    {
      const int tn = d ? (t - 1) : (t + 1);
      if (tn >= 0 && tn < Sq)
        __builtin_prefetch(G + ((long)(tn * Bq + mBase * 16) * (4 * HHq)) + kcol, 0, 0);
    }
    v8f acc[4][2];
#pragma unroll
    for (int g = 0; g < 4; ++g) { acc[g][0] = z; acc[g][1] = z; }
    for (int kk = 0; kk < HHq; kk += 32) {
      v16h a0 = load_a_frag(h_sh + ((mBase * 16) + (lane & 15)) * HHq + kk + selA);
      v16h a1 = load_a_frag(h_sh + ((mBase * 16 + 16) + (lane & 15)) * HHq + kk + selA);
#pragma unroll
      for (int g = 0; g < 4; ++g) {
        const int n = g * HHq + kcol;
        v16h bf = *(const v16h*)(whh + (long)n * HHq + kk + selB);
        acc[g][0] = wmma_f16(a0, bf, acc[g][0]);
        acc[g][1] = wmma_f16(a1, bf, acc[g][1]);
      }
    }
    __syncthreads();                 // all reads of h_sh done
#pragma unroll
    for (int mt = 0; mt < 2; ++mt)
#pragma unroll
      for (int r = 0; r < 8; ++r) {
        const int b = (mBase + mt) * 16 + ((lane & 16) ? r + 8 : r);
        const long row = (long)(t * Bq + b);
        const _Float16* g4 = G + row * (4 * HHq) + kcol;
        float gi = acc[0][mt][r] + (float)g4[0 * HHq];
        float gf = acc[1][mt][r] + (float)g4[1 * HHq];
        float gg = acc[2][mt][r] + (float)g4[2 * HHq];
        float go = acc[3][mt][r] + (float)g4[3 * HHq];
        float cn = sigm(gf) * c[mt][r] + sigm(gi) * tanhf(gg);
        float hn = sigm(go) * tanhf(cn);
        bool m = t < len[mt][r];
        if (m) { c[mt][r] = cn; h_sh[b * HHq + kcol] = (_Float16)hn; }
        bi[row * Hq + d * HHq + kcol] = (_Float16)(m ? hn : 0.0f);
      }
    __syncthreads();
  }
}

// ---------------------------------------------------------------------------
// Layer-2 recurrence: 2 stacked H=512 cells chained within each timestep.
// Gates C[64x2048]: wave w owns k-cols [w*16,w*16+16), n-tiles {w,w+32,w+64,
// w+96} (i,f,g,o). The batch (M) dimension is processed in two halves per
// step (m-tiles {0,1} then {2,3}) so accumulators stay at 64 VGPRs/wave.
// M-halves touch disjoint h rows -> GEMM(half); sync; epi(half) is race-free.
// l0 input proj precomputed (G0); l1 input proj (h0@Wih1^T) fused per step.
// ---------------------------------------------------------------------------
__global__ __launch_bounds__(1024) void k_lstm_l2(
    const _Float16* __restrict__ whh0_in, const _Float16* __restrict__ wih1_in,
    const _Float16* __restrict__ whh1_in, const _Float16* __restrict__ G0,
    const float* __restrict__ b_s,     // (2, 2048), layer-1 bias at +2048
    const float* __restrict__ h2_0,    // (2, 512)
    const float* __restrict__ c2_0,    // (2, 512)
    _Float16* __restrict__ outs) {
  extern __shared__ _Float16 sm[];
  _Float16* h0 = sm;                   // 64x512 fp16 (64 KB)
  _Float16* h1 = sm + Bq * Hq;         // 64x512 fp16 (64 KB)
  const int tid = threadIdx.x, wave = tid >> 5, lane = tid & 31;
  const int kcol = wave * 16 + (lane & 15);
  const int selA = (lane & 16) ? 16 : 0;
  const int selB = (lane & 16) ? 16 : 0;
  for (int i = tid; i < Bq * Hq; i += 1024) {
    h0[i] = (_Float16)h2_0[i & (Hq - 1)];
    h1[i] = (_Float16)h2_0[Hq + (i & (Hq - 1))];
  }
  float c0[4][8], c1v[4][8];
#pragma unroll
  for (int mt = 0; mt < 4; ++mt)
#pragma unroll
    for (int r = 0; r < 8; ++r) {
      c0[mt][r] = c2_0[kcol];
      c1v[mt][r] = c2_0[Hq + kcol];
    }
  __syncthreads();
  v8f z = {0.f, 0.f, 0.f, 0.f, 0.f, 0.f, 0.f, 0.f};
  int inv = 0;   // always 0; opaquely redefined per step (anti-LICM)
  for (int t = 0; t < Sq; ++t) {
    asm volatile("" : "+s"(inv));
    const _Float16* whh0 = whh0_in + inv;
    const _Float16* wih1 = wih1_in + inv;
    const _Float16* whh1 = whh1_in + inv;
    if (t + 1 < Sq)
      __builtin_prefetch(G0 + ((long)((t + 1) * Bq) * (4 * Hq)) + kcol, 0, 0);

    // ---- layer 0: gates = G0[t] + h0 @ whh0^T, two M-halves ----
#pragma unroll 1
    for (int mh = 0; mh < 2; ++mh) {
      v8f acc[4][2];
#pragma unroll
      for (int g = 0; g < 4; ++g) { acc[g][0] = z; acc[g][1] = z; }
      for (int kk = 0; kk < Hq; kk += 32) {
        v16h a0 = load_a_frag(h0 + ((mh * 2 + 0) * 16 + (lane & 15)) * Hq + kk + selA);
        v16h a1 = load_a_frag(h0 + ((mh * 2 + 1) * 16 + (lane & 15)) * Hq + kk + selA);
#pragma unroll
        for (int g = 0; g < 4; ++g) {
          const int n = g * Hq + kcol;
          v16h bf = *(const v16h*)(whh0 + (long)n * Hq + kk + selB);
          acc[g][0] = wmma_f16(a0, bf, acc[g][0]);
          acc[g][1] = wmma_f16(a1, bf, acc[g][1]);
        }
      }
      __syncthreads();   // this half's h0 reads complete
#pragma unroll
      for (int mi = 0; mi < 2; ++mi)
#pragma unroll
        for (int r = 0; r < 8; ++r) {
          const int mt = mh * 2 + mi;
          const int b = mt * 16 + ((lane & 16) ? r + 8 : r);
          const long row = (long)(t * Bq + b);
          const _Float16* g4 = G0 + row * (4 * Hq) + kcol;
          float gi = acc[0][mi][r] + (float)g4[0 * Hq];
          float gf = acc[1][mi][r] + (float)g4[1 * Hq];
          float gg = acc[2][mi][r] + (float)g4[2 * Hq];
          float go = acc[3][mi][r] + (float)g4[3 * Hq];
          float cn = sigm(gf) * c0[mt][r] + sigm(gi) * tanhf(gg);
          float hn = sigm(go) * tanhf(cn);
          c0[mt][r] = cn;
          h0[b * Hq + kcol] = (_Float16)hn;   // rows of THIS half only
        }
    }
    __syncthreads();

    // ---- layer 1: gates = h0 @ wih1^T + h1 @ whh1^T + b_s[1], two halves ----
#pragma unroll 1
    for (int mh = 0; mh < 2; ++mh) {
      v8f acc[4][2];
#pragma unroll
      for (int g = 0; g < 4; ++g) { acc[g][0] = z; acc[g][1] = z; }
      for (int kk = 0; kk < Hq; kk += 32) {
        v16h a0 = load_a_frag(h0 + ((mh * 2 + 0) * 16 + (lane & 15)) * Hq + kk + selA);
        v16h a1 = load_a_frag(h0 + ((mh * 2 + 1) * 16 + (lane & 15)) * Hq + kk + selA);
#pragma unroll
        for (int g = 0; g < 4; ++g) {
          const int n = g * Hq + kcol;
          v16h bf = *(const v16h*)(wih1 + (long)n * Hq + kk + selB);
          acc[g][0] = wmma_f16(a0, bf, acc[g][0]);
          acc[g][1] = wmma_f16(a1, bf, acc[g][1]);
        }
      }
      for (int kk = 0; kk < Hq; kk += 32) {
        v16h a0 = load_a_frag(h1 + ((mh * 2 + 0) * 16 + (lane & 15)) * Hq + kk + selA);
        v16h a1 = load_a_frag(h1 + ((mh * 2 + 1) * 16 + (lane & 15)) * Hq + kk + selA);
#pragma unroll
        for (int g = 0; g < 4; ++g) {
          const int n = g * Hq + kcol;
          v16h bf = *(const v16h*)(whh1 + (long)n * Hq + kk + selB);
          acc[g][0] = wmma_f16(a0, bf, acc[g][0]);
          acc[g][1] = wmma_f16(a1, bf, acc[g][1]);
        }
      }
      __syncthreads();   // this half's h0/h1 reads complete
#pragma unroll
      for (int mi = 0; mi < 2; ++mi)
#pragma unroll
        for (int r = 0; r < 8; ++r) {
          const int mt = mh * 2 + mi;
          const int b = mt * 16 + ((lane & 16) ? r + 8 : r);
          const long row = (long)(t * Bq + b);
          float gi = acc[0][mi][r] + b_s[2048 + 0 * Hq + kcol];
          float gf = acc[1][mi][r] + b_s[2048 + 1 * Hq + kcol];
          float gg = acc[2][mi][r] + b_s[2048 + 2 * Hq + kcol];
          float go = acc[3][mi][r] + b_s[2048 + 3 * Hq + kcol];
          float cn = sigm(gf) * c1v[mt][r] + sigm(gi) * tanhf(gg);
          float hn = sigm(go) * tanhf(cn);
          c1v[mt][r] = cn;
          h1[b * Hq + kcol] = (_Float16)hn;   // rows of THIS half only
          outs[row * Hq + kcol] = (_Float16)hn;
        }
    }
    __syncthreads();
  }
}

// ---------------------------------------------------------------------------
// Viterbi: one wave32 per batch row; trans + backpointers in LDS.
// ---------------------------------------------------------------------------
__global__ __launch_bounds__(32) void k_viterbi(const float* __restrict__ feats,
                                                const int* __restrict__ lengths,
                                                const float* __restrict__ trans,
                                                float* __restrict__ out) {
  const int b = blockIdx.x, lane = threadIdx.x;
  __shared__ float tr[Tq * Tq];
  __shared__ float sc[32];
  __shared__ unsigned char bp[Sq][Tq];  // bp[k] = vstep at t=k+1
  __shared__ int tags[Sq];
  for (int i = lane; i < Tq * Tq; i += 32) tr[i] = trans[i];
  const int len = lengths[b];
  __syncthreads();
  float score = -3.0e38f;
  if (lane < Tq) score = feats[(0 * Bq + b) * TPAD + lane] + tr[lane * Tq + START_TAG];
  for (int t = 1; t < Sq; ++t) {
    sc[lane] = score;
    __syncthreads();
    if (lane < Tq) {
      float best = -3.0e38f;
      int bj = 0;
      for (int j = 0; j < Tq; ++j) {
        float v = sc[j] + tr[lane * Tq + j];
        if (v > best) { best = v; bj = j; }
      }
      bool m = t < len;
      if (m) score = best + feats[(t * Bq + b) * TPAD + lane];
      bp[t - 1][lane] = (unsigned char)(m ? bj : lane);
    }
    __syncthreads();
  }
  sc[lane] = (lane < Tq) ? (score + tr[STOP_TAG * Tq + lane]) : -3.0e38f;
  __syncthreads();
  if (lane == 0) {
    float best = -3.0e38f;
    int bt = 0;
    for (int i = 0; i < Tq; ++i)
      if (sc[i] > best) { best = sc[i]; bt = i; }
    out[b] = best;
    tags[Sq - 1] = bt;
    for (int k = Sq - 2; k >= 0; --k) { bt = bp[k][bt]; tags[k] = bt; }
  }
  __syncthreads();
  for (int t = lane; t < Sq; t += 32) out[Bq + b * Sq + t] = (float)tags[t];
}

// ---------------------------------------------------------------------------
// launch
// ---------------------------------------------------------------------------
extern "C" void kernel_launch(void* const* d_in, const int* in_sizes, int n_in,
                              void* d_out, int out_size, void* d_ws, size_t ws_size,
                              hipStream_t stream) {
  const int*   sent  = (const int*)d_in[0];
  const int*   lens  = (const int*)d_in[1];
  const float* emb   = (const float*)d_in[2];
  const float* wih_f = (const float*)d_in[3];
  const float* whh_f = (const float*)d_in[4];
  const float* b_f   = (const float*)d_in[5];
  const float* wih_b = (const float*)d_in[6];
  const float* whh_b = (const float*)d_in[7];
  const float* b_b   = (const float*)d_in[8];
  const float* h1_0  = (const float*)d_in[9];
  const float* c1_0  = (const float*)d_in[10];
  const float* wih_s = (const float*)d_in[11];
  const float* whh_s = (const float*)d_in[12];
  const float* b_s   = (const float*)d_in[13];
  const float* h2_0  = (const float*)d_in[14];
  const float* c2_0  = (const float*)d_in[15];
  const float* W_tag = (const float*)d_in[16];
  const float* b_tag = (const float*)d_in[17];
  const float* trans = (const float*)d_in[18];
  (void)in_sizes; (void)n_in; (void)out_size; (void)ws_size;

  char* ws = (char*)d_ws;
  size_t o = 0;
  auto take = [&](size_t bytes) -> char* {
    char* p = ws + o;
    o = (o + bytes + 255) & ~(size_t)255;
    return p;
  };
  _Float16* wihF16 = (_Float16*)take(1024L * 256 * 2);
  _Float16* whhF16 = (_Float16*)take(1024L * 256 * 2);
  _Float16* wihB16 = (_Float16*)take(1024L * 256 * 2);
  _Float16* whhB16 = (_Float16*)take(1024L * 256 * 2);
  _Float16* wihS0  = (_Float16*)take(2048L * 512 * 2);
  _Float16* whhS0  = (_Float16*)take(2048L * 512 * 2);
  _Float16* wihS1  = (_Float16*)take(2048L * 512 * 2);
  _Float16* whhS1  = (_Float16*)take(2048L * 512 * 2);
  _Float16* wtag16 = (_Float16*)take((long)TPAD * 512 * 2);
  float*    btagP  = (float*)take(TPAD * 4);
  _Float16* x16    = (_Float16*)take((long)RROWS * 256 * 2);
  _Float16* Gf     = (_Float16*)take((long)RROWS * 1024 * 2);
  _Float16* Gb     = (_Float16*)take((long)RROWS * 1024 * 2);
  _Float16* G0     = Gf;  // alias: Gf+Gb (consumed by l1) == RROWS x 2048 f16
  _Float16* bi     = (_Float16*)take((long)RROWS * 512 * 2);
  _Float16* outs   = (_Float16*)take((long)RROWS * 512 * 2);
  float*    feats  = (float*)take((long)RROWS * TPAD * 4);

  // weights -> fp16
  k_cvt<<<1024, 256, 0, stream>>>(wih_f, wihF16, 262144);
  k_cvt<<<1024, 256, 0, stream>>>(whh_f, whhF16, 262144);
  k_cvt<<<1024, 256, 0, stream>>>(wih_b, wihB16, 262144);
  k_cvt<<<1024, 256, 0, stream>>>(whh_b, whhB16, 262144);
  k_cvt<<<4096, 256, 0, stream>>>(wih_s, wihS0, 1048576);
  k_cvt<<<4096, 256, 0, stream>>>(whh_s, whhS0, 1048576);
  k_cvt<<<4096, 256, 0, stream>>>(wih_s + 1048576, wihS1, 1048576);
  k_cvt<<<4096, 256, 0, stream>>>(whh_s + 1048576, whhS1, 1048576);
  k_wtag<<<64, 256, 0, stream>>>(W_tag, b_tag, wtag16, btagP);

  // embedding gather (time-major rows)
  k_gather<<<RROWS, 128, 0, stream>>>(sent, emb, x16);

  // layer-1 input projections (batched over all timesteps)
  k_gemm<128, false><<<dim3(1024 / 128, RROWS / 128), 256, 0, stream>>>(
      x16, wihF16, b_f, Gf, 1024, 256);
  k_gemm<128, false><<<dim3(1024 / 128, RROWS / 128), 256, 0, stream>>>(
      x16, wihB16, b_b, Gb, 1024, 256);

  // bidirectional recurrence (persistent, 2 blocks = 2 directions)
  k_lstm_l1<<<2, 1024, 0, stream>>>(whhF16, whhB16, Gf, Gb, h1_0, c1_0, lens, bi);

  // layer-2 bottom-cell input projection (bi fully known) -> overwrites Gf/Gb
  k_gemm<128, false><<<dim3(2048 / 128, RROWS / 128), 256, 0, stream>>>(
      bi, wihS0, b_s, G0, 2048, 512);

  // stacked 2-layer recurrence (persistent, l1 input proj fused per-step)
  k_lstm_l2<<<1, 1024, 2 * Bq * Hq * (int)sizeof(_Float16), stream>>>(
      whhS0, wihS1, whhS1, G0, b_s, h2_0, c2_0, outs);

  // tag projection (T padded to 32)
  k_gemm<32, true><<<dim3(1, RROWS / 128), 256, 0, stream>>>(
      outs, wtag16, btagP, feats, TPAD, 512);

  // Viterbi decode + backtrace, writes [scores(64) | tags(64x512)] as float
  k_viterbi<<<Bq, 32, 0, stream>>>(feats, lens, trans, (float*)d_out);
}
